// HierarchicalHopfield_455266533847
// MI455X (gfx1250) — matrix operand order repaired and checked by reading, not compiled, not run on hardware
//
#include <hip/hip_runtime.h>
#include <hip/hip_bf16.h>

// ---------------------------------------------------------------------------
// Problem constants (from reference)
// ---------------------------------------------------------------------------
#define BB   4096
#define DD   512
#define NCC  12
#define KCC  500
#define KCP  512      // padded class keys
#define KGG  5000
#define KGP  5120     // padded global keys (5 chunks of 1024)
#define HH   64
#define CHUNK 1024
#define TPB  256
#define WAVES 8

typedef __attribute__((ext_vector_type(16))) __bf16 v16bf;
typedef __attribute__((ext_vector_type(8)))  __bf16 v8bf;
typedef __attribute__((ext_vector_type(8)))  float  v8f;

union AF { v16bf v; v8bf h[2]; };

static __device__ __forceinline__ v8f wmma_bf16(const v16bf& a, const v16bf& b, const v8f& c) {
  // D = A(16x32) * B(32x16) + C ; emits v_wmma_f32_16x16x32_bf16
  return __builtin_amdgcn_wmma_f32_16x16x32_bf16(false, a, false, b, (short)0, c, false, false);
}

// A-fragment (16x32, M = lane&15). Per ISA: lanes 0-15 hold K {0..7}u{16..23},
// lanes 16-31 hold K {8..15}u{24..31} -> two contiguous 16B loads.
static __device__ __forceinline__ v16bf load_afrag(const __bf16* row_base, int k0, int hi) {
  AF a;
  a.h[0] = *(const v8bf*)(row_base + k0 + hi * 8);
  a.h[1] = *(const v8bf*)(row_base + k0 + 16 + hi * 8);
  return a.v;
}

// B-fragment (32x16, N = lane&15, row-major source with leading dim = K extent).
// Lane n holds column n; lanes 0-15 K {0..7}u{16..23}, lanes 16-31 K {8..15}u{24..31}.
static __device__ __forceinline__ v16bf load_bfrag(const __bf16* col_row_base, int k0, int hi) {
  AF b;
  const __bf16* p = col_row_base + k0 + hi * 8;
  b.h[0] = *(const v8bf*)(p);
  b.h[1] = *(const v8bf*)(p + 16);
  return b.v;
}

// ---------------------------------------------------------------------------
// Kernel 0: fp32 -> bf16 conversion, transposed copies, zero padding + cret=0
// ---------------------------------------------------------------------------
__global__ void __launch_bounds__(256) k_prep(
    const float* __restrict__ q, const float* __restrict__ gm, const float* __restrict__ cm,
    __bf16* __restrict__ qbf, __bf16* __restrict__ gbf, __bf16* __restrict__ gbfT,
    __bf16* __restrict__ cbf, __bf16* __restrict__ cbfT, float* __restrict__ cret)
{
  const int NQ  = BB * DD;          // 2,097,152
  const int NG  = KGP * DD;         // 2,621,440
  const int NGT = DD * KGP;
  const int NCB = NCC * KCP * DD;   // 3,145,728
  const int NCT = NCB;
  const int NZ  = BB * DD;
  const int TOT = NQ + NG + NGT + NCB + NCT + NZ;
  for (int i = blockIdx.x * TPB + threadIdx.x; i < TOT; i += gridDim.x * TPB) {
    int r = i;
    if (r < NQ) { qbf[r] = (__bf16)q[r]; continue; }
    r -= NQ;
    if (r < NG) {
      int row = r / DD;
      gbf[r] = (row < KGG) ? (__bf16)gm[r] : (__bf16)0.f;
      continue;
    }
    r -= NG;
    if (r < NGT) {
      int d = r / KGP, k = r % KGP;
      gbfT[r] = (k < KGG) ? (__bf16)gm[k * DD + d] : (__bf16)0.f;
      continue;
    }
    r -= NGT;
    if (r < NCB) {
      int c = r / (KCP * DD), rem = r % (KCP * DD);
      int k = rem / DD, d = rem % DD;
      cbf[r] = (k < KCC) ? (__bf16)cm[(c * KCC + k) * DD + d] : (__bf16)0.f;
      continue;
    }
    r -= NCB;
    if (r < NCT) {
      int c = r / (DD * KCP), rem = r % (DD * KCP);
      int d = rem / KCP, k = rem % KCP;
      cbfT[r] = (k < KCC) ? (__bf16)cm[(c * KCC + k) * DD + d] : (__bf16)0.f;
      continue;
    }
    r -= NCT;
    cret[r] = 0.f;
  }
}

// ---------------------------------------------------------------------------
// Kernel 1: global attention, flash-style online softmax.
// Block = 16 queries, 8 waves; wave w owns output columns [64w, 64w+64).
// ---------------------------------------------------------------------------
__global__ void __launch_bounds__(256) k_global_attn(
    const __bf16* __restrict__ qbf,   // BB x DD
    const __bf16* __restrict__ gbf,   // KGP x DD   (keys, row-major, zero padded)
    const __bf16* __restrict__ gbfT,  // DD x KGP   (values transposed)
    float* __restrict__ gret)         // BB x DD
{
  __shared__ __bf16 lds_q[16 * DD];       // 16 KB
  __shared__ __bf16 lds_p[16 * CHUNK];    // 32 KB scores -> probs (bf16)
  __shared__ float  red[TPB];
  __shared__ float  m_run[16], l_run[16], scl[16];

  const int tid  = threadIdx.x;
  const int wave = tid >> 5;
  const int lane = tid & 31;
  const int hi   = lane >> 4;
  const int lm   = lane & 15;
  const int b0   = blockIdx.x * 16;
  const int ncol0 = wave * 64;

  // stage Q tile into LDS (16B vector copies)
  {
    const v8bf* src = (const v8bf*)(qbf + (size_t)b0 * DD);
    v8bf* dst = (v8bf*)lds_q;
    for (int i = tid; i < 16 * DD / 8; i += TPB) dst[i] = src[i];
  }
  if (tid < 16) { m_run[tid] = -1e30f; l_run[tid] = 0.f; }
  __syncthreads();

  v8f Oa[4];
  for (int i = 0; i < 4; i++)
    for (int e = 0; e < 8; e++) Oa[i][e] = 0.f;

  for (int cbase = 0; cbase < KGP; cbase += CHUNK) {
    // ---- scores: 64 tiles of 16 keys, round-robin across 8 waves ----
    for (int t = wave; t < CHUNK / 16; t += WAVES) {
      const int n0 = cbase + t * 16;
      if (n0 + 128 < KGP)  // speculative prefetch of a later key tile
        __builtin_prefetch(gbf + (size_t)(n0 + 128 + lm) * DD, 0, 1);
      v8f acc;
      for (int e = 0; e < 8; e++) acc[e] = 0.f;
      const __bf16* qrow = lds_q + lm * DD;
      const __bf16* krow = gbf + (size_t)(n0 + lm) * DD;
      for (int ks = 0; ks < DD / 32; ks++) {
        v16bf a = load_afrag(qrow, ks * 32, hi);
        v16bf b = load_bfrag(krow, ks * 32, hi);
        acc = wmma_bf16(a, b, acc);
      }
      // store to LDS (C layout: lane half picks M offset, lane&15 = N)
      const int key   = n0 + lm;
      const bool ok   = key < KGG;
      const int  pc   = t * 16 + lm;
      for (int r = 0; r < 8; r++) {
        const int m = r + hi * 8;
        float s = ok ? acc[r] : -1e30f;
        lds_p[m * CHUNK + pc] = (__bf16)s;
      }
    }
    __syncthreads();

    // ---- online softmax over this chunk ----
    const int row = tid >> 4, sub = tid & 15;
    float mloc = -1e30f;
    for (int c = sub; c < CHUNK; c += 16)
      mloc = fmaxf(mloc, (float)lds_p[row * CHUNK + c]);
    red[tid] = mloc;
    __syncthreads();
    if (tid < 16) {
      float m = -1e30f;
      for (int i = 0; i < 16; i++) m = fmaxf(m, red[tid * 16 + i]);
      float mnew = fmaxf(m_run[tid], m);
      scl[tid]   = __expf(m_run[tid] - mnew);
      m_run[tid] = mnew;
    }
    __syncthreads();
    const float mnew = m_run[row];
    float psum = 0.f;
    for (int c = sub; c < CHUNK; c += 16) {
      float p = __expf((float)lds_p[row * CHUNK + c] - mnew);
      lds_p[row * CHUNK + c] = (__bf16)p;
      psum += p;
    }
    red[tid] = psum;
    __syncthreads();
    if (tid < 16) {
      float s = 0.f;
      for (int i = 0; i < 16; i++) s += red[tid * 16 + i];
      l_run[tid] = l_run[tid] * scl[tid] + s;
    }

    // ---- rescale accumulator, then P·V for this wave's 64 columns ----
    for (int i = 0; i < 4; i++)
      for (int r = 0; r < 8; r++) Oa[i][r] *= scl[r + hi * 8];

    for (int ks = 0; ks < CHUNK / 32; ks++) {
      const int kl = ks * 32;
      v16bf p = load_afrag(lds_p + lm * CHUNK, kl, hi);
      const int kg = cbase + kl;
      for (int i = 0; i < 4; i++) {
        const int nc = ncol0 + i * 16 + lm;               // output column (= gbfT row)
        v16bf b = load_bfrag(gbfT + (size_t)nc * KGP, kg, hi);
        Oa[i] = wmma_bf16(p, b, Oa[i]);
      }
    }
    __syncthreads();  // before next chunk overwrites lds_p
  }

  for (int r = 0; r < 8; r++) {
    const int m = r + hi * 8;
    const float inv = 1.f / l_run[m];
    for (int i = 0; i < 4; i++)
      gret[(size_t)(b0 + m) * DD + ncol0 + i * 16 + lm] = Oa[i][r] * inv;
  }
}

// ---------------------------------------------------------------------------
// Kernel 2: class attention, block = (16-query tile, class c). Early-exit if
// no row in the tile selected class c; weighted atomic accumulate of 0.5*r.
// ---------------------------------------------------------------------------
__global__ void __launch_bounds__(256) k_class_attn(
    const __bf16* __restrict__ qbf,
    const __bf16* __restrict__ cbf,   // NCC x KCP x DD
    const __bf16* __restrict__ cbfT,  // NCC x DD x KCP
    const int*    __restrict__ cls,   // BB x 2
    float*        __restrict__ cret)  // BB x DD (accumulated)
{
  __shared__ __bf16 lds_q[16 * DD];      // 16 KB
  __shared__ __bf16 lds_p[16 * KCP];     // 16 KB
  __shared__ float  red[TPB];
  __shared__ float  rowm[16], rowl[16], wgt[16];
  __shared__ int    anyflag;

  const int tid  = threadIdx.x;
  const int wave = tid >> 5;
  const int lane = tid & 31;
  const int hi   = lane >> 4;
  const int lm   = lane & 15;
  const int b0   = blockIdx.x * 16;
  const int c    = blockIdx.y;
  const int ncol0 = wave * 64;

  if (tid == 0) anyflag = 0;
  __syncthreads();
  if (tid < 16) {
    int i0 = cls[(b0 + tid) * 2 + 0];
    int i1 = cls[(b0 + tid) * 2 + 1];
    float w = 0.5f * (float)((i0 == c) + (i1 == c));
    wgt[tid] = w;
    if (w != 0.f) anyflag = 1;
  }
  __syncthreads();
  if (!anyflag) return;

  {
    const v8bf* src = (const v8bf*)(qbf + (size_t)b0 * DD);
    v8bf* dst = (v8bf*)lds_q;
    for (int i = tid; i < 16 * DD / 8; i += TPB) dst[i] = src[i];
  }
  __syncthreads();

  const __bf16* kmat = cbf  + (size_t)c * KCP * DD;
  const __bf16* vT   = cbfT + (size_t)c * DD * KCP;

  // ---- scores: 32 key tiles across 8 waves ----
  for (int t = wave; t < KCP / 16; t += WAVES) {
    const int n0 = t * 16;
    v8f acc;
    for (int e = 0; e < 8; e++) acc[e] = 0.f;
    const __bf16* qrow = lds_q + lm * DD;
    const __bf16* krow = kmat + (size_t)(n0 + lm) * DD;
    for (int ks = 0; ks < DD / 32; ks++) {
      v16bf a = load_afrag(qrow, ks * 32, hi);
      v16bf b = load_bfrag(krow, ks * 32, hi);
      acc = wmma_bf16(a, b, acc);
    }
    const bool ok = (n0 + lm) < KCC;
    for (int r = 0; r < 8; r++) {
      const int m = r + hi * 8;
      float s = ok ? acc[r] : -1e30f;
      lds_p[m * KCP + (n0 + lm)] = (__bf16)s;
    }
  }
  __syncthreads();

  // ---- softmax (single chunk) ----
  const int row = tid >> 4, sub = tid & 15;
  float mloc = -1e30f;
  for (int cc = sub; cc < KCP; cc += 16)
    mloc = fmaxf(mloc, (float)lds_p[row * KCP + cc]);
  red[tid] = mloc;
  __syncthreads();
  if (tid < 16) {
    float m = -1e30f;
    for (int i = 0; i < 16; i++) m = fmaxf(m, red[tid * 16 + i]);
    rowm[tid] = m;
  }
  __syncthreads();
  const float mrow = rowm[row];
  float psum = 0.f;
  for (int cc = sub; cc < KCP; cc += 16) {
    float p = __expf((float)lds_p[row * KCP + cc] - mrow);
    lds_p[row * KCP + cc] = (__bf16)p;
    psum += p;
  }
  red[tid] = psum;
  __syncthreads();
  if (tid < 16) {
    float s = 0.f;
    for (int i = 0; i < 16; i++) s += red[tid * 16 + i];
    rowl[tid] = s;
  }
  __syncthreads();

  // ---- P·V ----
  v8f Oa[4];
  for (int i = 0; i < 4; i++)
    for (int e = 0; e < 8; e++) Oa[i][e] = 0.f;
  for (int ks = 0; ks < KCP / 32; ks++) {
    const int kl = ks * 32;
    v16bf p = load_afrag(lds_p + lm * KCP, kl, hi);
    for (int i = 0; i < 4; i++) {
      const int nc = ncol0 + i * 16 + lm;
      v16bf b = load_bfrag(vT + (size_t)nc * KCP, kl, hi);
      Oa[i] = wmma_bf16(p, b, Oa[i]);
    }
  }

  for (int r = 0; r < 8; r++) {
    const int m = r + hi * 8;
    const float w = wgt[m];
    if (w != 0.f) {
      const float s = w / rowl[m];
      for (int i = 0; i < 4; i++)
        atomicAdd(&cret[(size_t)(b0 + m) * DD + ncol0 + i * 16 + lm], Oa[i][r] * s);
    }
  }
}

// ---------------------------------------------------------------------------
// Kernel 3: gating MLP (exact GELU) + blend. 64 threads per batch row.
// ---------------------------------------------------------------------------
__global__ void __launch_bounds__(256) k_gate(
    const float* __restrict__ cret, const float* __restrict__ gret,
    const float* __restrict__ w1, const float* __restrict__ b1,
    const float* __restrict__ w2, const float* __restrict__ b2,
    float* __restrict__ out)
{
  __shared__ float red[TPB];
  __shared__ float gates[4];
  const int tid = threadIdx.x;
  const int j = tid & 63;
  const int g = tid >> 6;
  const int bi = blockIdx.x * 4 + g;
  const float* cr = cret + (size_t)bi * DD;
  const float* gr = gret + (size_t)bi * DD;

  float h = b1[j];
  for (int t = 0; t < DD; t++) h += cr[t] * w1[t * HH + j];
  for (int t = 0; t < DD; t++) h += gr[t] * w1[(DD + t) * HH + j];
  float ge = 0.5f * h * (1.f + erff(h * 0.70710678118654752f));
  red[tid] = ge * w2[j];
  __syncthreads();
  if (j == 0) {
    float s = b2[0];
    for (int t = 0; t < 64; t++) s += red[g * 64 + t];
    gates[g] = 1.f / (1.f + __expf(-s));
  }
  __syncthreads();
  const float gate = gates[g];
  for (int d = j * 8; d < j * 8 + 8; d++)
    out[(size_t)bi * DD + d] = gate * cr[d] + (1.f - gate) * gr[d];
}

// ---------------------------------------------------------------------------
// Launcher
// ---------------------------------------------------------------------------
extern "C" void kernel_launch(void* const* d_in, const int* in_sizes, int n_in,
                              void* d_out, int out_size, void* d_ws, size_t ws_size,
                              hipStream_t stream) {
  const float* q   = (const float*)d_in[0];
  const int*   cls = (const int*)  d_in[1];
  const float* gm  = (const float*)d_in[2];
  const float* cm  = (const float*)d_in[3];
  const float* w1  = (const float*)d_in[4];
  const float* b1  = (const float*)d_in[5];
  const float* w2  = (const float*)d_in[6];
  const float* b2  = (const float*)d_in[7];
  float* out = (float*)d_out;

  const size_t SZ_QBF  = (size_t)BB * DD * 2;
  const size_t SZ_GBF  = (size_t)KGP * DD * 2;
  const size_t SZ_GBT  = (size_t)DD * KGP * 2;
  const size_t SZ_CBF  = (size_t)NCC * KCP * DD * 2;
  const size_t SZ_CBT  = SZ_CBF;
  const size_t SZ_GRET = (size_t)BB * DD * 4;
  const size_t SZ_CRET = SZ_GRET;
  const size_t NEED = SZ_QBF + SZ_GBF + SZ_GBT + SZ_CBF + SZ_CBT + SZ_GRET + SZ_CRET;
  if (ws_size < NEED) return;

  char* ws = (char*)d_ws;
  size_t off = 0;
  auto take = [&](size_t bytes) -> char* {
    char* p = ws + off;
    off += (bytes + 255) & ~(size_t)255;
    return p;
  };
  __bf16* qbf  = (__bf16*)take(SZ_QBF);
  __bf16* gbf  = (__bf16*)take(SZ_GBF);
  __bf16* gbfT = (__bf16*)take(SZ_GBT);
  __bf16* cbf  = (__bf16*)take(SZ_CBF);
  __bf16* cbfT = (__bf16*)take(SZ_CBT);
  float*  gret = (float*) take(SZ_GRET);
  float*  cret = (float*) take(SZ_CRET);

  k_prep<<<8192, TPB, 0, stream>>>(q, gm, cm, qbf, gbf, gbfT, cbf, cbfT, cret);
  k_global_attn<<<BB / 16, TPB, 0, stream>>>(qbf, gbf, gbfT, gret);
  dim3 g2(BB / 16, NCC);
  k_class_attn<<<g2, TPB, 0, stream>>>(qbf, cbf, cbfT, cls, cret);
  k_gate<<<BB / 4, TPB, 0, stream>>>(cret, gret, w1, b1, w2, b2, out);
}
